// SpMiddlePillarEncoder18_8942121910493
// MI455X (gfx1250) — matrix operand order repaired and checked by reading, not compiled, not run on hardware
//
#include <hip/hip_runtime.h>
#include <hip/hip_bf16.h>
#include <cstdint>

typedef __bf16 bf16;
typedef __attribute__((ext_vector_type(16))) __bf16 v16bf;
typedef __attribute__((ext_vector_type(8)))  float  v8f;

#define BN_EPS 1e-3f

// ---------------------------------------------------------------------------
// zero-fill (activation arena: establishes the zero halos once per launch)
// ---------------------------------------------------------------------------
__global__ void k_zero(uint4* __restrict__ p, long n16) {
  long i = (long)blockIdx.x * 256 + threadIdx.x;
  if (i < n16) p[i] = (uint4){0u, 0u, 0u, 0u};
}

// ---------------------------------------------------------------------------
// x (f32 NCHW) * mask -> bf16 NHWC with 1px zero halo; write level-1 u8 mask
// ---------------------------------------------------------------------------
__global__ void k_prep_input(const float* __restrict__ x, const int* __restrict__ mask,
                             bf16* __restrict__ out, unsigned char* __restrict__ m1,
                             int C, int H, int W, long total) {
  long i = (long)blockIdx.x * blockDim.x + threadIdx.x;
  if (i >= total) return;
  int  c  = (int)(i % C);
  long t  = i / C;
  int  yx = (int)(t % (H * W));
  int  b  = (int)(t / (H * W));
  int  y  = yx / W, xw = yx - (yx / W) * W;
  int  mk = mask[(long)b * H * W + yx] != 0;
  float v = mk ? x[((long)(b * C + c)) * H * W + yx] : 0.f;
  out[((long)(b * (H + 2) + y + 1) * (W + 2) + xw + 1) * C + c] = (bf16)v;
  if (c == 0) m1[(long)b * H * W + yx] = (unsigned char)mk;
}

// ---------------------------------------------------------------------------
// mask downsample: out(y,x) = OR over 3x3 window at stride 2 (pad 1)
// ---------------------------------------------------------------------------
__global__ void k_mask_down(const unsigned char* __restrict__ mi, unsigned char* __restrict__ mo,
                            int Bn, int Hi, int Wi) {
  int Ho = Hi >> 1, Wo = Wi >> 1;
  int i = blockIdx.x * 256 + threadIdx.x;
  int total = Bn * Ho * Wo;
  if (i >= total) return;
  int b = i / (Ho * Wo);
  int rem = i - b * Ho * Wo;
  int yo = rem / Wo, xo = rem - (rem / Wo) * Wo;
  unsigned char any = 0;
  for (int dy = -1; dy <= 1; ++dy)
    for (int dx = -1; dx <= 1; ++dx) {
      int iy = 2 * yo + dy, ix = 2 * xo + dx;
      if ((unsigned)iy < (unsigned)Hi && (unsigned)ix < (unsigned)Wi)
        any |= mi[((long)b * Hi + iy) * Wi + ix];
    }
  mo[i] = any ? 1 : 0;
}

// ---------------------------------------------------------------------------
// Fold BN into weights: wt[tap][cinChunk][Cout][32] bf16, bias[Cout] f32
// ---------------------------------------------------------------------------
__global__ void k_prep_weights(const float* __restrict__ w, const float* __restrict__ g,
                               const float* __restrict__ bt, const float* __restrict__ mn,
                               const float* __restrict__ vr, bf16* __restrict__ wt,
                               float* __restrict__ bias, int Cout, int Cin) {
  int i = blockIdx.x * 256 + threadIdx.x;
  int total = Cout * Cin * 9;
  if (i >= total) return;
  int co  = i / (Cin * 9);
  int rem = i - co * Cin * 9;
  int ci  = rem / 9;
  int tap = rem - ci * 9;
  float s = g[co] * rsqrtf(vr[co] + BN_EPS);
  int cc = ci >> 5, k = ci & 31;
  int cinChunks = Cin >> 5;
  wt[(((long)(tap * cinChunks + cc) * Cout + co) << 5) + k] = (bf16)(w[i] * s);
  if (ci == 0 && tap == 0) bias[co] = bt[co] - mn[co] * s;
}

// ---------------------------------------------------------------------------
// Implicit-GEMM 3x3 conv, bf16 WMMA, double-buffered LDS pipeline.
// Activations are halo-padded NHWC ((H+2)x(W+2), zero border), so every tap
// read is in-bounds: A staging = per-lane base + UNIFORM scalar offset, no
// border compares/cndmasks in the hot loop.  Weights [tap][cinChunk][Cout][32].
//   <8,2>: 128px x 32ch (Cout=32)
//   <4,2>:  64px x 64ch (Cout=64)
//   <4,4>:  64px x 128ch (Cout>=128)
// Epilogue staged through LDS: coalesced 16B residual loads + output stores.
// ---------------------------------------------------------------------------
template <int MTILES, int NACC>
__global__ __launch_bounds__(256) void k_conv3x3_wmma(
    const bf16* __restrict__ in, const bf16* __restrict__ wt,
    const float* __restrict__ bias, const bf16* __restrict__ res,
    const unsigned char* __restrict__ msk, bf16* __restrict__ out,
    int Hi, int Wi, int Cin, int Ho, int Wo, int Cout, int stride,
    int totalPix) {
  constexpr int WAVES_N = 8 / MTILES;
  constexpr int MT = MTILES * 16;
  constexpr int NT = WAVES_N * NACC * 16;
  constexpr int ASEG = (MT * 4) / 256;            // A 16B segments per thread
  constexpr int BSEG = (NT * 4 + 255) / 256;      // B 16B segments per thread
  constexpr int SA   = 2 * MT * 32 * 2;           // A double-buffer bytes
  constexpr int SB   = 2 * NT * 32 * 2;           // B double-buffer bytes
  constexpr int SEPI = MT * NT * 2;               // epilogue tile bytes
  constexpr int SMEM = (SA + SB > SEPI) ? (SA + SB) : SEPI;
  constexpr int NSEG8 = NT / 8;                   // 16B segs per output row
  constexpr int SH = (NSEG8 == 4) ? 2 : (NSEG8 == 8) ? 3 : 4;

  __shared__ __align__(16) unsigned char smem[SMEM];
  bf16* lA  = (bf16*)smem;                        // [2][MT*32]
  bf16* lB  = (bf16*)(smem + SA);                 // [2][NT*32]
  bf16* epi = (bf16*)smem;                        // [MT][NT] (aliases staging)

  const int tid  = threadIdx.x;
  const int lane = tid & 31;
  const int wv   = tid >> 5;
  const int p0   = blockIdx.x * MT;
  const int n0   = blockIdx.y * NT;
  const int mrow = (wv % MTILES) << 4;
  const int ncol = (wv / MTILES) * (NACC * 16);
  const int cinChunks = Cin >> 5;
  const int HWo = Ho * Wo;
  const int Wp  = Wi + 2;                         // padded input pitch
  const long wstride = (long)Cout << 5;           // weight elems per K-chunk row

  // --- per-thread A staging segments: (r,s) fixed; base at tap(0,0) ---
  const bf16* aPtr[ASEG];
  int aLds[ASEG];
#pragma unroll
  for (int k = 0; k < ASEG; ++k) {
    int idx = tid + k * 256;
    int r = idx >> 2, s = idx & 3;
    int p = p0 + r;
    int pc = p < totalPix ? p : 0;
    int b   = pc / HWo;
    int rem = pc - b * HWo;
    int yo  = rem / Wo;
    int xo  = rem - yo * Wo;
    aPtr[k] = in + ((long)(b * (Hi + 2) + yo * stride) * Wp + xo * stride) * Cin + (s << 3);
    aLds[k] = (r << 5) + (s << 3);
  }

  // --- per-thread B staging segments ---
  long bGOff[BSEG];
  int  bLds[BSEG];
  bool bAct[BSEG];
#pragma unroll
  for (int k = 0; k < BSEG; ++k) {
    int idx = tid + k * 256;
    bAct[k] = idx < NT * 4;
    int ic  = bAct[k] ? idx : 0;
    int n = ic >> 2, s = ic & 3;
    bGOff[k] = ((long)(n0 + n) << 5) + (s << 3);
    bLds[k]  = (n << 5) + (s << 3);
  }

  // --- incremental K-slice bookkeeping for the NEXT tile to stage ---
  int lcc = 0, lky = 0, lkx = 0;
  const bf16* wRow = wt;
  auto advance = [&]() {
    if (++lcc == cinChunks) { lcc = 0; if (++lkx == 3) { lkx = 0; ++lky; } }
    wRow += wstride;
  };
  auto loadTile = [&](int buf) {
    const int aoff = (lky * Wp + lkx) * Cin + (lcc << 5);  // uniform (scalar)
#pragma unroll
    for (int k = 0; k < ASEG; ++k) {
      uint4 v = *(const uint4*)(aPtr[k] + aoff);
      *(uint4*)(&lA[buf * (MT * 32) + aLds[k]]) = v;
    }
#pragma unroll
    for (int k = 0; k < BSEG; ++k) {
      if (bAct[k]) {
        uint4 wv4 = *(const uint4*)(wRow + bGOff[k]);
        *(uint4*)(&lB[buf * (NT * 32) + bLds[k]]) = wv4;
      }
    }
  };

  v8f acc[NACC];
#pragma unroll
  for (int j = 0; j < NACC; ++j) acc[j] = (v8f){0.f, 0.f, 0.f, 0.f, 0.f, 0.f, 0.f, 0.f};

  // CDNA5 bf16 fragment addressing (A 16x32, B 32x16 VGPR layouts)
  const int aOff  = ((mrow + (lane & 15)) << 5) + ((lane >> 4) << 3);
  const int bOffB = ((ncol + (lane & 15)) << 5) + ((lane >> 4) << 3);

  const int T = 9 * cinChunks;
  loadTile(0);
  advance();
  __syncthreads();

  for (int it = 0; it < T; ++it) {
    const int buf = it & 1;
    if (it + 1 < T) { loadTile(buf ^ 1); advance(); }

    union U { uint4 u[2]; v16bf v; };
    U af;
    af.u[0] = *(const uint4*)(&lA[buf * (MT * 32) + aOff]);
    af.u[1] = *(const uint4*)(&lA[buf * (MT * 32) + aOff + 16]);
#pragma unroll
    for (int j = 0; j < NACC; ++j) {
      U bfr;
      int bo = buf * (NT * 32) + bOffB + ((j * 16) << 5);
      bfr.u[0] = *(const uint4*)(&lB[bo]);
      bfr.u[1] = *(const uint4*)(&lB[bo + 16]);
      acc[j] = __builtin_amdgcn_wmma_f32_16x16x32_bf16(false, af.v, false, bfr.v,
                                                       (short)0, acc[j], false, false);
    }
    __syncthreads();
  }

  // --- epilogue pass 1: acc + bias -> LDS tile [MT][NT] (bf16) ---
  const int hs = lane >> 4;
  const int nl = lane & 15;
#pragma unroll
  for (int j = 0; j < NACC; ++j) {
    float bz = bias[n0 + ncol + j * 16 + nl];
    int cl = ncol + j * 16 + nl;
#pragma unroll
    for (int r = 0; r < 8; ++r) {
      int m = mrow + r + hs * 8;
      epi[m * NT + cl] = (bf16)(acc[j][r] + bz);
    }
  }
  __syncthreads();

  // --- epilogue pass 2: coalesced relu(x + res) * mask into padded NHWC ---
  for (int i = tid; i < MT * NSEG8; i += 256) {
    int m  = i >> SH;
    int so = i & (NSEG8 - 1);
    int p  = p0 + m;
    if (p >= totalPix) continue;
    int b   = p / HWo;
    int rem = p - b * HWo;
    int yo  = rem / Wo;
    int xo  = rem - yo * Wo;
    float mv = msk ? (msk[p] ? 1.f : 0.f) : 1.f;
    long gidx = ((long)(b * (Ho + 2) + yo + 1) * (Wo + 2) + xo + 1) * Cout + n0 + (so << 3);
    union V { uint4 u; bf16 h[8]; };
    V a;
    a.u = *(const uint4*)(&epi[m * NT + (so << 3)]);
    if (res) {
      V rr;
      rr.u = *(const uint4*)(res + gidx);
#pragma unroll
      for (int e = 0; e < 8; ++e) {
        float f = (float)a.h[e] + (float)rr.h[e];
        f = f > 0.f ? f : 0.f;
        a.h[e] = (bf16)(f * mv);
      }
    } else {
#pragma unroll
      for (int e = 0; e < 8; ++e) {
        float f = (float)a.h[e];
        f = f > 0.f ? f : 0.f;
        a.h[e] = (bf16)(f * mv);
      }
    }
    *(uint4*)(out + gidx) = a.u;
  }
}

// ---------------------------------------------------------------------------
// padded bf16 NHWC -> f32 NCHW output conversion
// ---------------------------------------------------------------------------
__global__ void k_to_nchw(const bf16* __restrict__ in, float* __restrict__ out,
                          int Bn, int C, int H, int W) {
  long i = (long)blockIdx.x * 256 + threadIdx.x;
  long total = (long)Bn * C * H * W;
  if (i >= total) return;
  int  hw = (int)(i % (H * W));
  long t  = i / (H * W);
  int  c  = (int)(t % C);
  int  b  = (int)(t / C);
  int  y  = hw / W, xw = hw - (hw / W) * W;
  out[i] = (float)in[((long)(b * (H + 2) + y + 1) * (W + 2) + xw + 1) * C + c];
}

// ---------------------------------------------------------------------------
// Host orchestration
// Input leaf order assumes jax tree flattening (sorted dict keys):
//   0: mask | 1..115: params (see index table) | 116: x
// Workspace need ~280 MB (halo-padded activations + weights + masks).
// ---------------------------------------------------------------------------
extern "C" void kernel_launch(void* const* d_in, const int* in_sizes, int n_in,
                              void* d_out, int out_size, void* d_ws, size_t ws_size,
                              hipStream_t stream) {
  (void)in_sizes; (void)n_in; (void)out_size; (void)ws_size;
  const int Bn = 4;

  char* wsb = (char*)d_ws;
  size_t off = 0;
  auto alloc = [&](size_t bytes) -> void* {
    off = (off + 255) & ~(size_t)255;
    void* p = wsb + off;
    off += bytes;
    return p;
  };
  auto padElems = [&](int H, int W, int C) -> size_t {
    return (size_t)Bn * (H + 2) * (W + 2) * C;
  };

  bf16* S0 = (bf16*)alloc(padElems(512, 512, 32) * 2);
  bf16* S1 = (bf16*)alloc(padElems(512, 512, 32) * 2);
  bf16* o1 = (bf16*)alloc(padElems(512, 512, 32) * 2);
  bf16* o2 = (bf16*)alloc(padElems(256, 256, 64) * 2);
  bf16* o3 = (bf16*)alloc(padElems(128, 128, 128) * 2);
  bf16* o4 = (bf16*)alloc(padElems(64, 64, 256) * 2);
  bf16* o5 = (bf16*)alloc(padElems(32, 32, 256) * 2);
  size_t actEnd = off;                           // zero everything up to here
  unsigned char* m1 = (unsigned char*)alloc((size_t)Bn * 512 * 512);
  unsigned char* m2 = (unsigned char*)alloc((size_t)Bn * 256 * 256);
  unsigned char* m3 = (unsigned char*)alloc((size_t)Bn * 128 * 128);
  unsigned char* m4 = (unsigned char*)alloc((size_t)Bn * 64 * 64);

  // zero-fill the activation arena (establishes halos; pads never rewritten)
  {
    long n16 = (long)((actEnd + 15) / 16);
    k_zero<<<(int)((n16 + 255) / 256), 256, 0, stream>>>((uint4*)d_ws, n16);
  }

  struct WB { bf16* w; float* b; };
  auto allocW = [&](int Cout, int Cin) -> WB {
    WB r;
    r.w = (bf16*)alloc((size_t)9 * Cin * Cout * 2);
    r.b = (float*)alloc((size_t)Cout * 4);
    return r;
  };
  auto F = [&](int i) -> const float* { return (const float*)d_in[i]; };
  auto prepw = [&](int wi, int bni, WB wb, int Cout, int Cin) {
    int tot = Cout * Cin * 9;
    k_prep_weights<<<(tot + 255) / 256, 256, 0, stream>>>(
        F(wi), F(bni), F(bni + 1), F(bni + 2), F(bni + 3), wb.w, wb.b, Cout, Cin);
  };

  // --- weights (sorted-key flatten index table) ---
  WB w_s1[5];  const int s1w[5]  = {13, 14, 15, 24, 25};
               const int s1bn[5] = { 1,  5,  9, 16, 20};
  for (int i = 0; i < 5; ++i) { w_s1[i] = allocW(32, 32);  prepw(s1w[i], s1bn[i], w_s1[i], 32, 32); }
  WB w_s2d = allocW(64, 32);   prepw(50, 46, w_s2d, 64, 32);
  WB w_s2[4];  const int s2w[4]  = {34, 35, 44, 45};
               const int s2bn[4] = {26, 30, 36, 40};
  for (int i = 0; i < 4; ++i) { w_s2[i] = allocW(64, 64);  prepw(s2w[i], s2bn[i], w_s2[i], 64, 64); }
  WB w_s3d = allocW(128, 64);  prepw(75, 71, w_s3d, 128, 64);
  WB w_s3[4];  const int s3w[4]  = {59, 60, 69, 70};
               const int s3bn[4] = {51, 55, 61, 65};
  for (int i = 0; i < 4; ++i) { w_s3[i] = allocW(128, 128); prepw(s3w[i], s3bn[i], w_s3[i], 128, 128); }
  WB w_s4d = allocW(256, 128); prepw(100, 96, w_s4d, 256, 128);
  WB w_s4[4];  const int s4w[4]  = {84, 85, 94, 95};
               const int s4bn[4] = {76, 80, 86, 90};
  for (int i = 0; i < 4; ++i) { w_s4[i] = allocW(256, 256); prepw(s4w[i], s4bn[i], w_s4[i], 256, 256); }
  WB w_s5[3];  const int s5w[3]  = {101, 106, 111};
               const int s5bn[3] = {102, 107, 112};
  for (int i = 0; i < 3; ++i) { w_s5[i] = allocW(256, 256); prepw(s5w[i], s5bn[i], w_s5[i], 256, 256); }

  // --- input + mask pyramid ---
  {
    long total = (long)Bn * 512 * 512 * 32;
    k_prep_input<<<(int)((total + 255) / 256), 256, 0, stream>>>(
        F(116), (const int*)d_in[0], S0, m1, 32, 512, 512, total);
  }
  auto mdown = [&](const unsigned char* mi, unsigned char* mo, int Hi, int Wi) {
    int tot = Bn * (Hi / 2) * (Wi / 2);
    k_mask_down<<<(tot + 255) / 256, 256, 0, stream>>>(mi, mo, Bn, Hi, Wi);
  };
  mdown(m1, m2, 512, 512);
  mdown(m2, m3, 256, 256);
  mdown(m3, m4, 128, 128);

  auto conv = [&](const bf16* in, WB wb, const bf16* res, const unsigned char* msk,
                  bf16* outp, int Hi, int Wi, int Cin, int Cout, int stride) {
    int Ho = Hi / stride, Wo = Wi / stride;
    int total = Bn * Ho * Wo;
    if (Cout == 32) {
      dim3 grid((total + 127) / 128, 1);
      k_conv3x3_wmma<8, 2><<<grid, 256, 0, stream>>>(in, wb.w, wb.b, res, msk, outp,
          Hi, Wi, Cin, Ho, Wo, Cout, stride, total);
    } else if (Cout == 64) {
      dim3 grid((total + 63) / 64, 1);
      k_conv3x3_wmma<4, 2><<<grid, 256, 0, stream>>>(in, wb.w, wb.b, res, msk, outp,
          Hi, Wi, Cin, Ho, Wo, Cout, stride, total);
    } else {
      dim3 grid((total + 63) / 64, Cout / 128);
      k_conv3x3_wmma<4, 4><<<grid, 256, 0, stream>>>(in, wb.w, wb.b, res, msk, outp,
          Hi, Wi, Cin, Ho, Wo, Cout, stride, total);
    }
  };

  // stage 1 (512x512, C32, mask m1)
  conv(S0, w_s1[0], nullptr, m1, S1, 512, 512, 32, 32, 1);   // blockV conv0
  conv(S1, w_s1[1], nullptr, m1, o1, 512, 512, 32, 32, 1);   // b0.conv1
  conv(o1, w_s1[2], S1,      m1, S0, 512, 512, 32, 32, 1);   // b0.conv2 + res
  conv(S0, w_s1[3], nullptr, m1, S1, 512, 512, 32, 32, 1);   // b1.conv1
  conv(S1, w_s1[4], S0,      m1, o1, 512, 512, 32, 32, 1);   // b1.conv2 + res -> x1
  // stage 2 (down to 256x256, C64, mask m2)
  conv(o1, w_s2d,  nullptr, m2, S0, 512, 512, 32, 64, 2);
  conv(S0, w_s2[0], nullptr, m2, S1, 256, 256, 64, 64, 1);
  conv(S1, w_s2[1], S0,      m2, S0, 256, 256, 64, 64, 1);
  conv(S0, w_s2[2], nullptr, m2, S1, 256, 256, 64, 64, 1);
  conv(S1, w_s2[3], S0,      m2, o2, 256, 256, 64, 64, 1);   // -> x2
  // stage 3 (down to 128x128, C128, mask m3)
  conv(o2, w_s3d,  nullptr, m3, S0, 256, 256, 64, 128, 2);
  conv(S0, w_s3[0], nullptr, m3, S1, 128, 128, 128, 128, 1);
  conv(S1, w_s3[1], S0,      m3, S0, 128, 128, 128, 128, 1);
  conv(S0, w_s3[2], nullptr, m3, S1, 128, 128, 128, 128, 1);
  conv(S1, w_s3[3], S0,      m3, o3, 128, 128, 128, 128, 1); // -> x3
  // stage 4 (down to 64x64, C256, mask m4)
  conv(o3, w_s4d,  nullptr, m4, S0, 128, 128, 128, 256, 2);
  conv(S0, w_s4[0], nullptr, m4, S1, 64, 64, 256, 256, 1);
  conv(S1, w_s4[1], S0,      m4, S0, 64, 64, 256, 256, 1);
  conv(S0, w_s4[2], nullptr, m4, S1, 64, 64, 256, 256, 1);
  conv(S1, w_s4[3], S0,      m4, o4, 64, 64, 256, 256, 1);   // -> x4
  // stage 5 (dense, down to 32x32)
  conv(o4, w_s5[0], nullptr, nullptr, S0, 64, 64, 256, 256, 2);
  conv(S0, w_s5[1], nullptr, nullptr, S1, 32, 32, 256, 256, 1);
  conv(S1, w_s5[2], nullptr, nullptr, o5, 32, 32, 256, 256, 1); // -> h

  // outputs: padded NHWC bf16 -> NCHW f32 concatenated
  float* outp = (float*)d_out;
  auto tonchw = [&](const bf16* src, float* dst, int C, int H, int W) {
    long tot = (long)Bn * C * H * W;
    k_to_nchw<<<(int)((tot + 255) / 256), 256, 0, stream>>>(src, dst, Bn, C, H, W);
  };
  tonchw(o1, outp,              32, 512, 512);
  tonchw(o2, outp + 33554432L,  64, 256, 256);
  tonchw(o3, outp + 50331648L, 128, 128, 128);
  tonchw(o4, outp + 58720256L, 256, 64, 64);
  tonchw(o5, outp + 62914560L, 256, 32, 32);
}
